// Attention_37082747633811
// MI455X (gfx1250) — matrix operand order repaired and checked by reading, not compiled, not run on hardware
//
#include <hip/hip_runtime.h>
#include <hip/hip_bf16.h>
#include <cstdint>

// ---------------------------------------------------------------------------
// Attention on gfx1250 (MI455X): f16 WMMA (v_wmma_f32_16x16x32_f16) for all
// three GEMMs, f32 accumulation, two-pass softmax in 263KB LDS per workgroup.
// Round 3: 32x64 register tiles, double-buffered fragments (verified: rolled
// loop with staged s_wait_loadcnt), WGP-scope prefetch (locality 3 so the
// global_prefetch_b8 pulls into WGP$, not SCOPE_SYS/GL2-only).
// ---------------------------------------------------------------------------

typedef _Float16 half16 __attribute__((ext_vector_type(16)));
typedef _Float16 half8  __attribute__((ext_vector_type(8)));
typedef float    float8 __attribute__((ext_vector_type(8)));
typedef float    f32x4  __attribute__((ext_vector_type(4)));  // 'float4' taken by HIP

union H16 { half16 v; half8 h[2]; };

#define DIM_   768
#define HEADS_ 12
#define HD_    64
#define BATCH_ 8
#define SEQ_   1024
#define ROWS_  (BATCH_ * SEQ_)     // 8192
#define QKVC_  (3 * DIM_)          // 2304
#define SCALE_ 0.03608439182435161f  // DIM^-0.5  (full dim, per reference)
#define SROW_  1028                // padded LDS row stride (floats)

static __device__ __forceinline__
float8 wmma16(half16 a, half16 b, float8 c) {
  // D = A(16x32) * B(32x16) + C(16x16 f32)
  return __builtin_amdgcn_wmma_f32_16x16x32_f16(
      /*neg_a=*/false, a, /*neg_b=*/false, b,
      /*c_mod=*/(short)0, c, /*reuse_a=*/false, /*reuse_b=*/false);
}

// A fragment: lane m = lane&15 is the row; kb = (lane>=16)?8:0.
// elements 0..15 = A[m][k+2*kb+0..15].
static __device__ __forceinline__
half16 load_a(const _Float16* __restrict__ row, int k, int kb) {
  H16 u;
  u.h[0] = *(const half8*)(row + k + kb * 2);
  u.h[1] = *(const half8*)(row + k + kb * 2 + 8);
  return u.v;
}

// B fragment: lane n = lane&15 is the column; koff = (lane>=16)?16:0.
// elements 0..15 = B[k+koff+0..15][n]  == Bt[n][k+koff .. +15] (Bt row-major).
static __device__ __forceinline__
half16 load_b(const _Float16* __restrict__ btrow, int k, int koff) {
  return *(const half16*)(btrow + k + koff);
}

// ---------------------------------------------------------------------------
// Elementwise conversions
// ---------------------------------------------------------------------------
__global__ void k_cvt_f32_f16(const float* __restrict__ src,
                              _Float16* __restrict__ dst, int n) {
  int i = blockIdx.x * 256 + threadIdx.x;
  if (i < n) dst[i] = (_Float16)src[i];
}

// dst[c][r] = src[r][c]  (f32 -> f16 transpose; makes weights B-operand ready)
__global__ void k_transpose_cvt(const float* __restrict__ src,
                                _Float16* __restrict__ dst,
                                int rows, int cols) {
  int i = blockIdx.x * 256 + threadIdx.x;
  if (i < rows * cols) {
    int r = i / cols, c = i % cols;
    dst[(size_t)c * rows + r] = (_Float16)src[i];
  }
}

// ---------------------------------------------------------------------------
// QKV projection: [8192,768] x [768,2304], scatter to Q/K (row-major per
// head) and V transposed ([bh][d][n]) in f16.
// One wave computes a 32x64 tile (2 A-frags, 8 accumulators, B shared).
// grid = (8192/32)*(2304/64)/8 waves = 256*36/8 = 1152 blocks of 256.
// ---------------------------------------------------------------------------
__global__ void k_qkv_gemm(const _Float16* __restrict__ X,
                           const _Float16* __restrict__ Wt,   // [2304][768]
                           _Float16* __restrict__ Q,
                           _Float16* __restrict__ K,
                           _Float16* __restrict__ Vt) {
  const int lane = threadIdx.x & 31;
  const int wave = threadIdx.x >> 5;
  const int id   = blockIdx.x * 8 + wave;       // 0 .. 9215
  const int nt   = id % 36;                     // 64-wide N tile
  const int mt   = id / 36;                     // 32-wide M tile (<256)

  const int m    = lane & 15;
  const int kb   = (lane >> 4) << 3;            // 0 or 8
  const int koff = (lane >> 4) << 4;            // 0 or 16
  const int hi8  = (lane >> 4) << 3;

  const _Float16* arow0 = X + (size_t)(mt * 32 + m) * DIM_;
  const _Float16* arow1 = arow0 + (size_t)16 * DIM_;
  const _Float16* brow[4];
  #pragma unroll
  for (int j = 0; j < 4; ++j)
    brow[j] = Wt + (size_t)(nt * 64 + j * 16 + m) * DIM_;

  float8 acc[2][4] = {};

  // ---- software-pipelined k loop: load k+32 fragments before k's WMMAs ----
  half16 ca0 = load_a(arow0, 0, kb);
  half16 ca1 = load_a(arow1, 0, kb);
  half16 cb[4];
  #pragma unroll
  for (int j = 0; j < 4; ++j) cb[j] = load_b(brow[j], 0, koff);

  for (int k = 0; k < DIM_; k += 32) {
    int kn = (k + 32 < DIM_) ? k + 32 : 0;      // last-iter loads discarded
    __builtin_prefetch(arow0 + k + 256, 0, 3);  // WGP-scope, ~8 k-steps ahead
    __builtin_prefetch(brow[0] + k + 256, 0, 3);
    half16 na0 = load_a(arow0, kn, kb);
    half16 na1 = load_a(arow1, kn, kb);
    half16 nb[4];
    #pragma unroll
    for (int j = 0; j < 4; ++j) nb[j] = load_b(brow[j], kn, koff);

    #pragma unroll
    for (int j = 0; j < 4; ++j) {
      acc[0][j] = wmma16(ca0, cb[j], acc[0][j]);
      acc[1][j] = wmma16(ca1, cb[j], acc[1][j]);
    }
    ca0 = na0; ca1 = na1;
    #pragma unroll
    for (int j = 0; j < 4; ++j) cb[j] = nb[j];
  }

  #pragma unroll
  for (int sub = 0; sub < 2; ++sub) {
    const int row0 = mt * 32 + sub * 16 + hi8;
    #pragma unroll
    for (int j = 0; j < 4; ++j) {
      int c    = nt * 64 + j * 16;     // tile lies inside one (qkv, head)
      int qkvi = c / DIM_;
      int cc   = c % DIM_;
      int h    = cc / HD_;
      int d    = (cc % HD_) + m;
      #pragma unroll
      for (int r = 0; r < 8; ++r) {
        int grow = row0 + r;
        int b    = grow >> 10;
        int n    = grow & 1023;
        int bh   = b * HEADS_ + h;
        _Float16 v = (_Float16)acc[sub][j][r];
        if      (qkvi == 0) Q [((size_t)bh * SEQ_ + n) * HD_ + d] = v;
        else if (qkvi == 1) K [((size_t)bh * SEQ_ + n) * HD_ + d] = v;
        else                Vt[((size_t)bh * HD_ + d) * SEQ_ + n] = v;
      }
    }
  }
}

// ---------------------------------------------------------------------------
// Attention: grid (96, 16), block 128 (4 waves). Each wave owns 16 query rows
// of one (b,h); workgroup owns 64 rows. Scores for the full 1024 keys live in
// dynamic LDS (64 x 1028 f32 = 263KB, CDNA5 320KB/WGP). Two-pass softmax,
// then O = P*V via WMMA, normalized, stored f16 as [b*n][h*64+d].
// Only 1 wave/SIMD at this LDS footprint -> fragments double-buffered so
// global loads overlap the WMMA pipe.
// ---------------------------------------------------------------------------
__global__ void k_attn(const _Float16* __restrict__ Q,
                       const _Float16* __restrict__ K,
                       const _Float16* __restrict__ Vt,
                       _Float16* __restrict__ O) {
  extern __shared__ float smem[];               // [64][SROW_]
  __shared__ float pm[128], ps[128], rs[64];

  const int tid  = threadIdx.x;                 // 0..127
  const int lane = tid & 31;
  const int wave = tid >> 5;                    // 0..3
  const int bh   = blockIdx.x;                  // 0..95
  const int qb   = blockIdx.y;                  // 0..15 (64-row q block)
  const int b    = bh / HEADS_;
  const int h    = bh % HEADS_;

  const int m    = lane & 15;
  const int kb   = (lane >> 4) << 3;
  const int koff = (lane >> 4) << 4;
  const int hi8  = (lane >> 4) << 3;

  float* Sw = smem + (size_t)wave * 16 * SROW_; // this wave's 16 score rows

  // ---- Phase 1: S = (Q K^T) * SCALE ------------------------------------
  const _Float16* qbase = Q + ((size_t)bh * SEQ_ + qb * 64 + wave * 16) * HD_;
  const _Float16* kbase = K + (size_t)bh * SEQ_ * HD_;

  half16 aq0 = load_a(qbase + m * HD_, 0, kb);   // K-dim 0..31
  half16 aq1 = load_a(qbase + m * HD_, 32, kb);  // K-dim 32..63

  half16 cb0 = load_b(kbase + (size_t)m * HD_, 0, koff);
  half16 cb1 = load_b(kbase + (size_t)m * HD_, 32, koff);

  for (int jt = 0; jt < 64; ++jt) {
    int jn = (jt + 1) & 63;                     // last-iter loads discarded
    const _Float16* nk = kbase + (size_t)(jn * 16 + m) * HD_;
    half16 nb0 = load_b(nk, 0, koff);
    half16 nb1 = load_b(nk, 32, koff);

    float8 acc = {};
    acc = wmma16(aq0, cb0, acc);
    acc = wmma16(aq1, cb1, acc);

    int col = jt * 16 + m;
    #pragma unroll
    for (int r = 0; r < 8; ++r)
      Sw[(size_t)(hi8 + r) * SROW_ + col] = acc[r] * SCALE_;

    cb0 = nb0; cb1 = nb1;
  }
  __syncthreads();

  // ---- Phase 2: softmax over each row of 1024 (2 threads per row) ------
  {
    int row = tid >> 1;
    int c0  = (tid & 1) * 512;
    float* Pr = smem + (size_t)row * SROW_ + c0;
    float mx = -3.4e38f;
    for (int i = 0; i < 512; ++i) mx = fmaxf(mx, Pr[i]);
    pm[tid] = mx;
    __syncthreads();
    mx = fmaxf(pm[tid & ~1], pm[tid | 1]);
    float sum = 0.f;
    for (int i = 0; i < 512; ++i) {
      float e = __expf(Pr[i] - mx);
      Pr[i] = e;
      sum += e;
    }
    ps[tid] = sum;
    __syncthreads();
    if ((tid & 1) == 0) rs[row] = ps[tid] + ps[tid + 1];
  }
  __syncthreads();

  // ---- Phase 3: O = P V, normalize, store -------------------------------
  const _Float16* vbase = Vt + (size_t)bh * HD_ * SEQ_;
  float8 oacc[4] = {};
  const float* prow = Sw + (size_t)m * SROW_;

  half16 vb[4];
  #pragma unroll
  for (int dt = 0; dt < 4; ++dt)
    vb[dt] = load_b(vbase + (size_t)(dt * 16 + m) * SEQ_, 0, koff);

  for (int s = 0; s < 32; ++s) {
    int k  = s * 32;
    int kn = (s + 1 < 32) ? k + 32 : 0;         // last-iter loads discarded
    half16 nvb[4];
    #pragma unroll
    for (int dt = 0; dt < 4; ++dt)
      nvb[dt] = load_b(vbase + (size_t)(dt * 16 + m) * SEQ_, kn, koff);

    // A-frag: convert this wave's P rows (f32 in LDS) -> f16, b128 reads
    H16 ap;
    const float* pr = prow + k + 2 * kb;        // 16B-aligned (SROW_%4==0)
    f32x4 p0 = *(const f32x4*)(pr);
    f32x4 p1 = *(const f32x4*)(pr + 4);
    f32x4 p2 = *(const f32x4*)(pr + 8);
    f32x4 p3 = *(const f32x4*)(pr + 12);
    #pragma unroll
    for (int i = 0; i < 4; ++i) {
      ap.v[i]      = (_Float16)p0[i];
      ap.v[4 + i]  = (_Float16)p1[i];
      ap.v[8 + i]  = (_Float16)p2[i];
      ap.v[12 + i] = (_Float16)p3[i];
    }

    #pragma unroll
    for (int dt = 0; dt < 4; ++dt)
      oacc[dt] = wmma16(ap.v, vb[dt], oacc[dt]);

    #pragma unroll
    for (int dt = 0; dt < 4; ++dt) vb[dt] = nvb[dt];
  }

  #pragma unroll
  for (int dt = 0; dt < 4; ++dt) {
    #pragma unroll
    for (int r = 0; r < 8; ++r) {
      int mrow = hi8 + r;                            // 0..15
      int nq   = qb * 64 + wave * 16 + mrow;
      float val = oacc[dt][r] / rs[wave * 16 + mrow];
      O[((size_t)b * SEQ_ + nq) * DIM_ + h * HD_ + dt * 16 + m] = (_Float16)val;
    }
  }
}

// ---------------------------------------------------------------------------
// Output projection: [8192,768] x [768,768] + bias -> f32 out.
// One wave computes a 32x64 tile. grid = 256*12/8 = 384 blocks of 256.
// ---------------------------------------------------------------------------
__global__ void k_outproj(const _Float16* __restrict__ A,   // [8192][768]
                          const _Float16* __restrict__ Wt,  // [768][768] (N,K)
                          const float* __restrict__ bias,
                          float* __restrict__ out) {
  const int lane = threadIdx.x & 31;
  const int wave = threadIdx.x >> 5;
  const int id   = blockIdx.x * 8 + wave;       // 0..3071
  const int nt   = id % 12;
  const int mt   = id / 12;                     // <256

  const int m    = lane & 15;
  const int kb   = (lane >> 4) << 3;
  const int koff = (lane >> 4) << 4;
  const int hi8  = (lane >> 4) << 3;

  const _Float16* arow0 = A + (size_t)(mt * 32 + m) * DIM_;
  const _Float16* arow1 = arow0 + (size_t)16 * DIM_;
  const _Float16* brow[4];
  #pragma unroll
  for (int j = 0; j < 4; ++j)
    brow[j] = Wt + (size_t)(nt * 64 + j * 16 + m) * DIM_;

  float8 acc[2][4] = {};

  half16 ca0 = load_a(arow0, 0, kb);
  half16 ca1 = load_a(arow1, 0, kb);
  half16 cb[4];
  #pragma unroll
  for (int j = 0; j < 4; ++j) cb[j] = load_b(brow[j], 0, koff);

  for (int k = 0; k < DIM_; k += 32) {
    int kn = (k + 32 < DIM_) ? k + 32 : 0;
    __builtin_prefetch(arow0 + k + 256, 0, 3);
    __builtin_prefetch(brow[0] + k + 256, 0, 3);
    half16 na0 = load_a(arow0, kn, kb);
    half16 na1 = load_a(arow1, kn, kb);
    half16 nb[4];
    #pragma unroll
    for (int j = 0; j < 4; ++j) nb[j] = load_b(brow[j], kn, koff);

    #pragma unroll
    for (int j = 0; j < 4; ++j) {
      acc[0][j] = wmma16(ca0, cb[j], acc[0][j]);
      acc[1][j] = wmma16(ca1, cb[j], acc[1][j]);
    }
    ca0 = na0; ca1 = na1;
    #pragma unroll
    for (int j = 0; j < 4; ++j) cb[j] = nb[j];
  }

  #pragma unroll
  for (int sub = 0; sub < 2; ++sub) {
    const int row0 = mt * 32 + sub * 16 + hi8;
    #pragma unroll
    for (int j = 0; j < 4; ++j) {
      int col  = nt * 64 + j * 16 + m;
      float bv = bias[col];
      #pragma unroll
      for (int r = 0; r < 8; ++r)
        out[(size_t)(row0 + r) * DIM_ + col] = acc[sub][j][r] + bv;
    }
  }
}

// ---------------------------------------------------------------------------
extern "C" void kernel_launch(void* const* d_in, const int* in_sizes, int n_in,
                              void* d_out, int out_size, void* d_ws, size_t ws_size,
                              hipStream_t stream) {
  const float* x      = (const float*)d_in[0];   // [8,1024,768]
  const float* w_qkv  = (const float*)d_in[1];   // [768,2304]
  const float* w_out  = (const float*)d_in[2];   // [768,768]
  const float* b_out  = (const float*)d_in[3];   // [768]
  float* out          = (float*)d_out;           // [8,1024,768]

  char* ws = (char*)d_ws;
  size_t off = 0;
  auto carve = [&](size_t bytes) {
    char* p = ws + off;
    off += (bytes + 255) & ~(size_t)255;
    return p;
  };
  _Float16* Xf   = (_Float16*)carve((size_t)ROWS_ * DIM_ * 2);    // 12.6 MB
  _Float16* WqT  = (_Float16*)carve((size_t)QKVC_ * DIM_ * 2);    //  3.5 MB
  _Float16* WoT  = (_Float16*)carve((size_t)DIM_ * DIM_ * 2);     //  1.2 MB
  _Float16* Qb   = (_Float16*)carve((size_t)BATCH_ * HEADS_ * SEQ_ * HD_ * 2);
  _Float16* Kb   = (_Float16*)carve((size_t)BATCH_ * HEADS_ * SEQ_ * HD_ * 2);
  _Float16* Vt   = (_Float16*)carve((size_t)BATCH_ * HEADS_ * SEQ_ * HD_ * 2);
  _Float16* Ob   = (_Float16*)carve((size_t)ROWS_ * DIM_ * 2);
  (void)ws_size; (void)n_in; (void)in_sizes; (void)out_size;

  // 1) precision conversion + weight transposition (B-operand layout)
  k_cvt_f32_f16<<<(ROWS_ * DIM_) / 256, 256, 0, stream>>>(x, Xf, ROWS_ * DIM_);
  k_transpose_cvt<<<(DIM_ * QKVC_) / 256, 256, 0, stream>>>(w_qkv, WqT, DIM_, QKVC_);
  k_transpose_cvt<<<(DIM_ * DIM_) / 256, 256, 0, stream>>>(w_out, WoT, DIM_, DIM_);

  // 2) QKV projection (WMMA), scatter into attention-ready layouts
  k_qkv_gemm<<<(256 * 36) / 8, 256, 0, stream>>>(Xf, WqT, Qb, Kb, Vt);

  // 3) attention (WMMA + big-LDS two-pass softmax)
  k_attn<<<dim3(BATCH_ * HEADS_, SEQ_ / 64), 128,
           (size_t)64 * SROW_ * sizeof(float), stream>>>(Qb, Kb, Vt, Ob);

  // 4) output projection (WMMA) + bias
  k_outproj<<<(256 * 12) / 8, 256, 0, stream>>>(Ob, WoT, b_out, out);
}
